// GCNModel_79439715107081
// MI455X (gfx1250) — matrix operand order repaired and checked by reading, not compiled, not run on hardware
//
#include <hip/hip_runtime.h>
#include <hip/hip_bf16.h>

// GCN 2-layer forward for MI455X (gfx1250, wave32).
// GEMMs: v_wmma_f32_16x16x32_f16 (K=64 -> 2 WMMA per 16x16 tile, f32 acc).
// Propagation: self-loop init + per-edge gather/scale + global f32 atomics
// (both 25.6MB operands are L2-resident on the 192MB L2 -> atomic-rate bound).

typedef __attribute__((ext_vector_type(16))) _Float16 v16h;
typedef __attribute__((ext_vector_type(8)))  float    v8f;

#define NN   100000            // nodes
#define NPAD 100096            // 782 * 128: padded rows so GEMM stores are unguarded
#define NE   1600000           // edges
#define DIM  64
#define NG   128               // graphs

// ---------------- degree / norm ----------------
__global__ __launch_bounds__(256) void k_deg_init(float* __restrict__ deg, int n) {
  int i = blockIdx.x * 256 + threadIdx.x;
  if (i < n) deg[i] = 1.0f;                      // self-loop
}

__global__ __launch_bounds__(256) void k_deg_edges(const int* __restrict__ ei,
                                                   float* __restrict__ deg, int ne) {
  int e = blockIdx.x * 256 + threadIdx.x;
  if (e < ne) atomicAdd(&deg[ei[ne + e]], 1.0f); // in-degree on col (targets)
}

__global__ __launch_bounds__(256) void k_dinv(float* __restrict__ deg, int n) {
  int i = blockIdx.x * 256 + threadIdx.x;
  if (i < n) deg[i] = __frsqrt_rn(deg[i]);       // deg -> d^{-1/2} in place
}

// ---------------- WMMA GEMM: C[NPAD x 64] = act(A + bias) @ W ----------------
// One wave -> 16 rows x 64 cols (8 WMMAs). W is staged into LDS pre-packed in
// B-fragment order so each fragment is two contiguous ds_load_b128 per lane.
// C rows [NN, NPAD) are scratch padding -> stores are unconditional.
__global__ __launch_bounds__(256) void k_gemm_wmma(
    const float* __restrict__ A, const float* __restrict__ W,
    const float* __restrict__ bias, int fuse_bias_relu,
    float* __restrict__ C, int n_rows)
{
  // Packed B fragments: chunk index = (tile*2 + khalf)*32 + lane, 16 halves each.
  // Element j of chunk (t,h,l) = W[k][n], k = h*32 + (l>>4)*8 + j + (j>=8 ? 8 : 0), n = t*16 + (l&15).
  __shared__ _Float16 Wl[DIM * DIM];             // 8 KB of 320 KB WGP LDS
  {
    int tid = threadIdx.x;                       // 256 threads == 256 chunks
    int t = tid >> 6, h = (tid >> 5) & 1, l = tid & 31;
    int kbase = h * 32 + ((l >> 4) * 8);
    int ncol  = t * 16 + (l & 15);
#pragma unroll
    for (int j = 0; j < 16; ++j)
      Wl[tid * 16 + j] = (_Float16)W[(kbase + j + (j >= 8 ? 8 : 0)) * DIM + ncol];
  }
  __syncthreads();

  int tid  = threadIdx.x;
  int wave = tid >> 5, lane = tid & 31;
  int rowbase = blockIdx.x * 128 + wave * 16;
  int r = rowbase + (lane & 15);
  if (r >= n_rows) r = n_rows - 1;               // read clamp only (x is exact-size)
  int kb = (lane >> 4) * 8;                      // ISA 16-bit A-layout K base

  // A fragments: a0 = K 0..31, a1 = K 32..63, via float4 loads.
  const float4* ap4 = (const float4*)(A + (size_t)r * DIM);
  float va[32];
#pragma unroll
  for (int s = 0; s < 4; ++s) {                  // segments kb+{0,16,32,48}..+8
    float4 u0 = ap4[(kb + 16 * s) >> 2];
    float4 u1 = ap4[((kb + 16 * s) >> 2) + 1];
    va[s * 8 + 0] = u0.x; va[s * 8 + 1] = u0.y; va[s * 8 + 2] = u0.z; va[s * 8 + 3] = u0.w;
    va[s * 8 + 4] = u1.x; va[s * 8 + 5] = u1.y; va[s * 8 + 6] = u1.z; va[s * 8 + 7] = u1.w;
  }
  if (fuse_bias_relu) {                          // uniform branch (SGPR arg)
#pragma unroll
    for (int s = 0; s < 4; ++s)
#pragma unroll
      for (int j = 0; j < 8; ++j)
        va[s * 8 + j] = fmaxf(va[s * 8 + j] + bias[kb + 16 * s + j], 0.f);
  }
  v16h a0, a1;
#pragma unroll
  for (int j = 0; j < 8; ++j) {
    a0[j]     = (_Float16)va[j];                 // K = kb + j
    a0[8 + j] = (_Float16)va[8 + j];             // K = kb + 16 + j
    a1[j]     = (_Float16)va[16 + j];            // K = kb + 32 + j
    a1[8 + j] = (_Float16)va[24 + j];            // K = kb + 48 + j
  }

  const v16h* Wp = (const v16h*)Wl;
  int n = lane & 15;
#pragma unroll
  for (int t = 0; t < 4; ++t) {                  // 4 N-tiles of 16
    v16h b0 = Wp[(t * 2 + 0) * 32 + lane];
    v16h b1 = Wp[(t * 2 + 1) * 32 + lane];
    v8f acc = {};
    acc = __builtin_amdgcn_wmma_f32_16x16x32_f16(false, a0, false, b0,
                                                 (short)0, acc, false, false);
    acc = __builtin_amdgcn_wmma_f32_16x16x32_f16(false, a1, false, b1,
                                                 (short)0, acc, false, false);
    float* cp = C + (size_t)rowbase * DIM + t * 16 + n;
    int moff = (lane < 16) ? 0 : 8;              // C layout: VGPR v -> M=v / v+8
#pragma unroll
    for (int v = 0; v < 8; ++v)                  // unconditional (C padded to NPAD)
      cp[(size_t)(moff + v) * DIM] = acc[v];
  }
}

// ---------------- propagation ----------------
// p[i] = dinv[i]^2 * t[i]  (self-loop term; doubles as accumulator init)
__global__ __launch_bounds__(256) void k_self(const float* __restrict__ t,
                                              const float* __restrict__ dinv,
                                              float* __restrict__ p, int n)
{
  int gw = (blockIdx.x * 256 + threadIdx.x) >> 5;
  int lane = threadIdx.x & 31;
  if (gw >= n) return;
  float w = dinv[gw]; w *= w;
  float2 v = ((const float2*)(t + (size_t)gw * DIM))[lane];
  ((float2*)(p + (size_t)gw * DIM))[lane] = make_float2(v.x * w, v.y * w);
}

// one wave per edge: p[col] += dinv[row]*dinv[col] * t[row]
__global__ __launch_bounds__(256) void k_edge(const int* __restrict__ ei,
                                              const float* __restrict__ t,
                                              const float* __restrict__ dinv,
                                              float* __restrict__ p, int ne)
{
  int gw = (blockIdx.x * 256 + threadIdx.x) >> 5;
  int lane = threadIdx.x & 31;
  if (gw >= ne) return;
  int r = __builtin_amdgcn_readfirstlane(ei[gw]);       // wave-uniform -> SGPR
  int c = __builtin_amdgcn_readfirstlane(ei[ne + gw]);
  float w = dinv[r] * dinv[c];
  float2 v = ((const float2*)(t + (size_t)r * DIM))[lane];
  float* d = p + (size_t)c * DIM + lane * 2;
  atomicAdd(d,     v.x * w);
  atomicAdd(d + 1, v.y * w);
}

// ---------------- pooling + head ----------------
__global__ void k_zero_pool(float* __restrict__ sums, float* __restrict__ cnts) {
  for (int j = threadIdx.x; j < NG * DIM; j += 256) sums[j] = 0.f;
  if (threadIdx.x < NG) cnts[threadIdx.x] = 0.f;
}

// wave per node: sums[batch[i]] += relu(p[i]+b2), cnts[batch[i]] += 1
__global__ __launch_bounds__(256) void k_pool(const float* __restrict__ p,
                                              const float* __restrict__ b,
                                              const int* __restrict__ batch,
                                              float* __restrict__ sums,
                                              float* __restrict__ cnts, int n)
{
  int gw = (blockIdx.x * 256 + threadIdx.x) >> 5;
  int lane = threadIdx.x & 31;
  if (gw >= n) return;
  int g = __builtin_amdgcn_readfirstlane(batch[gw]);
  float2 v = ((const float2*)(p + (size_t)gw * DIM))[lane];
  float h0 = fmaxf(v.x + b[lane * 2],     0.f);
  float h1 = fmaxf(v.y + b[lane * 2 + 1], 0.f);
  atomicAdd(&sums[g * DIM + lane * 2],     h0);
  atomicAdd(&sums[g * DIM + lane * 2 + 1], h1);
  if (lane == 0) atomicAdd(&cnts[g], 1.0f);
}

// wave per graph: out[g] = (sums[g]/max(cnt,1)) . Wlin + blin
__global__ __launch_bounds__(256) void k_final(const float* __restrict__ sums,
                                               const float* __restrict__ cnts,
                                               const float* __restrict__ Wlin,
                                               const float* __restrict__ blin,
                                               float* __restrict__ out)
{
  int gw = (blockIdx.x * 256 + threadIdx.x) >> 5;
  int lane = threadIdx.x & 31;
  if (gw >= NG) return;
  float rc = __frcp_rn(fmaxf(cnts[gw], 1.0f));
  float s0 = sums[gw * DIM + lane * 2];
  float s1 = sums[gw * DIM + lane * 2 + 1];
  float part = (s0 * Wlin[lane * 2] + s1 * Wlin[lane * 2 + 1]) * rc;
#pragma unroll
  for (int off = 16; off > 0; off >>= 1)
    part += __shfl_down(part, off, 32);
  if (lane == 0) out[gw] = part + blin[0];
}

// ---------------- launch ----------------
extern "C" void kernel_launch(void* const* d_in, const int* in_sizes, int n_in,
                              void* d_out, int out_size, void* d_ws, size_t ws_size,
                              hipStream_t stream)
{
  (void)in_sizes; (void)n_in; (void)out_size; (void)ws_size;
  const float* x     = (const float*)d_in[0];
  const int*   ei    = (const int*)d_in[1];   // [2, NE] int32 (harness convention)
  const int*   batch = (const int*)d_in[2];
  const float* W1    = (const float*)d_in[3];
  const float* b1    = (const float*)d_in[4];
  const float* W2    = (const float*)d_in[5];
  const float* b2    = (const float*)d_in[6];
  const float* Wlin  = (const float*)d_in[7];
  const float* blin  = (const float*)d_in[8];
  float* out = (float*)d_out;

  char* ws = (char*)d_ws;
  float* dinv = (float*)ws; ws += (((size_t)NN * 4) + 255) / 256 * 256;
  float* bufA = (float*)ws; ws += (size_t)NPAD * DIM * 4;   // padded activations
  float* bufB = (float*)ws; ws += (size_t)NPAD * DIM * 4;
  float* sums = (float*)ws; ws += (size_t)NG * DIM * 4;
  float* cnts = (float*)ws;

  int nb_nodes = (NN + 255) / 256;
  int nb_edges = (NE + 255) / 256;
  int nb_gemm  = NPAD / 128;         // 782 full blocks, 8 waves x 16 rows each
  int nb_wnode = (NN + 7) / 8;       // wave per node
  int nb_wedge = (NE + 7) / 8;       // wave per edge

  // normalization
  k_deg_init <<<nb_nodes, 256, 0, stream>>>(dinv, NN);
  k_deg_edges<<<nb_edges, 256, 0, stream>>>(ei, dinv, NE);
  k_dinv     <<<nb_nodes, 256, 0, stream>>>(dinv, NN);

  // layer 1: t1 = x @ W1 ; p1 = Ahat t1
  k_gemm_wmma<<<nb_gemm, 256, 0, stream>>>(x, W1, nullptr, 0, bufA, NN);
  k_self     <<<nb_wnode, 256, 0, stream>>>(bufA, dinv, bufB, NN);
  k_edge     <<<nb_wedge, 256, 0, stream>>>(ei, bufA, dinv, bufB, NE);

  // layer 2: t2 = relu(p1 + b1) @ W2 (activation fused) ; p2 = Ahat t2
  k_gemm_wmma<<<nb_gemm, 256, 0, stream>>>(bufB, W2, b1, 1, bufA, NPAD);
  k_self     <<<nb_wnode, 256, 0, stream>>>(bufA, dinv, bufB, NN);
  k_edge     <<<nb_wedge, 256, 0, stream>>>(ei, bufA, dinv, bufB, NE);

  // relu(p2 + b2) fused into pooling; then per-graph head
  k_zero_pool<<<1, 256, 0, stream>>>(sums, cnts);
  k_pool     <<<nb_wnode, 256, 0, stream>>>(bufB, b2, batch, sums, cnts, NN);
  k_final    <<<(NG * 32 + 255) / 256, 256, 0, stream>>>(sums, cnts, Wlin, blin, out);
}
